// AttentionLayer_49984829391158
// MI455X (gfx1250) — matrix operand order repaired and checked by reading, not compiled, not run on hardware
//
#include <hip/hip_runtime.h>

typedef __attribute__((ext_vector_type(16))) _Float16 v16h;
typedef __attribute__((ext_vector_type(8)))  float    v8f;

#define LRELU_ALPHA 0.2f
#define DIM 256  // DW == DA == 256 in the reference

// ---------------------------------------------------------------------------
// Kernel 1: zero the [B, V] output with vectorized stores (float4)
// ---------------------------------------------------------------------------
__global__ void zero_f4_kernel(float4* __restrict__ out, long long n4,
                               float* __restrict__ tail, int ntail) {
  long long i = (long long)blockIdx.x * blockDim.x + threadIdx.x;
  if (i < n4) out[i] = make_float4(0.f, 0.f, 0.f, 0.f);
  if (i == 0) {
    for (int t = 0; t < ntail; ++t) tail[t] = 0.f;
  }
}

// ---------------------------------------------------------------------------
// Kernel 2/3: out[r] = dot(mat[r, 0:256], vec[0:256])
// via v_wmma_f32_16x16x32_f16, one wave32 per 16 rows.
//
// Operand mapping (CDNA5 ISA §7.12.2, wave32):
//   B (32x16 f16): lane n<16 -> column N=n holds K=0..15 (16 contiguous
//                  elements of table row r0+n); lane>=16 -> column N=n-16,
//                  K=16..31. => each lane does 4x global_load_b128, coalesced.
//   A (16x32 f16): vector lives in row M=0 only: lane 0 holds K=0..15,
//                  lane 16 holds K=16..31; all other lanes zero.
//                  Built ONCE before the K-loop, branchlessly (cndmask).
//   D (16x16 f32): D[M=0][N=n] = acc[0] in lane n (n=0..15) = result row n.
// ---------------------------------------------------------------------------
__global__ __launch_bounds__(256) void wmma_matvec_kernel(
    const float* __restrict__ mat,   // [nrows, 256] row-major f32
    const float* __restrict__ vec,   // [256] f32
    float* __restrict__ out,         // [nrows] f32
    int nrows)
{
  const int lane = threadIdx.x & 31;
  const int wave = threadIdx.x >> 5;
  const int tile = blockIdx.x * (blockDim.x >> 5) + wave;
  const int r0   = tile * 16;
  if (r0 >= nrows) return;                 // wave-uniform: EXEC stays all-1s

  const int  sub = lane & 15;
  const bool hi  = (lane >= 16);

  // ---- Build the 8 A-operands (vector in row M=0) once, branchless ----
  v16h Aop[8];
#pragma unroll
  for (int c = 0; c < 8; ++c) {
    const int kA = c * 32 + (hi ? 16 : 0);
#pragma unroll
    for (int i = 0; i < 8; ++i) {
      _Float16 lo = (_Float16)vec[kA + i];
      _Float16 hi16 = (_Float16)vec[kA + 8 + i];
      Aop[c][i]     = (sub == 0) ? lo   : (_Float16)0.0f;
      Aop[c][8 + i] = (sub == 0) ? hi16 : (_Float16)0.0f;
    }
  }

  // ---- Stream 16 table rows as the B operand; this lane owns column `sub`
  const int  brow  = r0 + sub;
  const bool valid = (brow < nrows);       // tail tile: clamp to in-bounds row
  const float4* bptr = (const float4*)(mat + (long long)(valid ? brow : r0) * DIM
                                           + (hi ? 16 : 0));

  v8f acc = {};
#pragma unroll
  for (int c = 0; c < 8; ++c) {
    // 16 contiguous floats = 4x float4 (global_load_b128)
    const float4 x0 = bptr[c * 8 + 0];
    const float4 x1 = bptr[c * 8 + 1];
    const float4 x2 = bptr[c * 8 + 2];
    const float4 x3 = bptr[c * 8 + 3];
    v16h Bop;
    Bop[0]  = (_Float16)x0.x; Bop[1]  = (_Float16)x0.y;
    Bop[2]  = (_Float16)x0.z; Bop[3]  = (_Float16)x0.w;
    Bop[4]  = (_Float16)x1.x; Bop[5]  = (_Float16)x1.y;
    Bop[6]  = (_Float16)x1.z; Bop[7]  = (_Float16)x1.w;
    Bop[8]  = (_Float16)x2.x; Bop[9]  = (_Float16)x2.y;
    Bop[10] = (_Float16)x2.z; Bop[11] = (_Float16)x2.w;
    Bop[12] = (_Float16)x3.x; Bop[13] = (_Float16)x3.y;
    Bop[14] = (_Float16)x3.z; Bop[15] = (_Float16)x3.w;
    acc = __builtin_amdgcn_wmma_f32_16x16x32_f16(
        /*neg_a=*/false, Aop[c], /*neg_b=*/false, Bop,
        /*c_mod=*/(short)0, acc, /*reuse_a=*/false, /*reuse_b=*/false);
  }

  // D row M=0: lane n (n<16) holds result for table row r0+n in acc[0]
  if (!hi && valid) out[brow] = acc[0];
}

// ---------------------------------------------------------------------------
// Kernel 4: e = leaky_relu(s[w] + attr_dot[b]); attn[b, w] = e
// Duplicate (b,w) pairs write identical values, matching .at[].set semantics.
// ---------------------------------------------------------------------------
__global__ void scatter_kernel(const int* __restrict__ words,
                               const float* __restrict__ s,
                               const float* __restrict__ attr_dot,
                               float* __restrict__ attn,
                               int L, int V, int total) {
  int idx = blockIdx.x * blockDim.x + threadIdx.x;
  if (idx >= total) return;
  const int b = idx / L;
  const int w = words[idx];
  float e = s[w] + attr_dot[b];
  e = (e > 0.f) ? e : LRELU_ALPHA * e;
  attn[(long long)b * V + w] = e;
}

// ---------------------------------------------------------------------------
extern "C" void kernel_launch(void* const* d_in, const int* in_sizes, int n_in,
                              void* d_out, int out_size, void* d_ws, size_t ws_size,
                              hipStream_t stream) {
  const int*   words    = (const int*)  d_in[0];  // [B, L]
  const float* table    = (const float*)d_in[1];  // [V, 256]
  const float* attr_emb = (const float*)d_in[2];  // [B, 256]
  const float* a_vec    = (const float*)d_in[3];  // [512, 1]
  float*       attn     = (float*)d_out;          // [B, V]

  const int BL = in_sizes[0];                     // B*L
  const int V  = in_sizes[1] / DIM;               // 50257
  const int B  = in_sizes[2] / DIM;               // 256
  const int L  = BL / B;                          // 512

  // Workspace: s[V] then attr_dot[B]
  float* s_tab    = (float*)d_ws;
  float* attr_dot = s_tab + ((V + 63) & ~63);

  // 1) zero the output
  {
    const long long n  = (long long)out_size;
    const long long n4 = n / 4;
    const int ntail    = (int)(n - n4 * 4);
    const int blocks   = (int)((n4 + 255) / 256);
    zero_f4_kernel<<<blocks, 256, 0, stream>>>((float4*)attn, n4,
                                               attn + n4 * 4, ntail);
  }
  // 2) s[v] = dot(table[v], a[0:256])  — streams the whole table once
  {
    const int tiles  = (V + 15) / 16;
    const int blocks = (tiles + 7) / 8;            // 8 waves / 256-thread block
    wmma_matvec_kernel<<<blocks, 256, 0, stream>>>(table, a_vec, s_tab, V);
  }
  // 3) attr_dot[b] = dot(attr_emb[b], a[256:512])
  {
    const int tiles  = (B + 15) / 16;
    const int blocks = (tiles + 7) / 8;
    wmma_matvec_kernel<<<blocks, 256, 0, stream>>>(attr_emb, a_vec + DIM,
                                                   attr_dot, B);
  }
  // 4) gather s / add / leaky-relu / scatter
  {
    const int blocks = (BL + 255) / 256;
    scatter_kernel<<<blocks, 256, 0, stream>>>(words, s_tab, attr_dot, attn,
                                               L, V, BL);
  }
}
